// GVPTransformer_29618094473613
// MI455X (gfx1250) — matrix operand order repaired and checked by reading, not compiled, not run on hardware
//
#include <hip/hip_runtime.h>
#include <cstdint>
#include <cstddef>

// ---------------------------------------------------------------------------
// GVP transformer forward for MI455X (gfx1250, wave32, WMMA).
// Heavy contractions use a f16-WMMA GEMM (v_wmma_f32_16x16x32_f16, fp32
// accumulate) with a 16x128 wave tile: A fragment converted once per k-step
// and reused across 8 WMMAs; weights pre-converted to compact f16 once per
// layer. Edge vector features are algebraically collapsed to rank-1
// (coef[e,c] * ev_unit[e,d]) so no (E,128,3) tensors are materialized.
// Workspace use ~370 MB.
// ---------------------------------------------------------------------------

typedef __attribute__((ext_vector_type(16))) _Float16 v16h;
typedef __attribute__((ext_vector_type(8)))  float    v8f;

#define LAY   3
#define NSC   128
#define NVC   128
#define ECC   128
#define NETC  5
#define NGC   123          // ECC - NETC
#define KNB   24           // neighbors
#define RMAXC 10.0f
#define WPB   8            // waves per block in GEMM

// ===========================================================================
// GEMM:  C[M x 128] = epi( A[M x K] @ W16^T + bias ),  W16 is [128 x K] f16
// compact row-major. A is fp32 (lda row stride), converted to f16 in-register.
// One wave produces a 16(M) x 128(N) strip: 8 accumulators, 8 WMMA / k-step.
// EPI: 0 = none, 1 = leaky_relu(slope), 2 = sigmoid.
// A/B k order: contiguous, k = 16*(lane/16) + s in both fragments (wave32).
// ===========================================================================
__device__ __forceinline__ v16h pack16(float4 p0, float4 p1, float4 p2, float4 p3) {
  v16h r;
  r[0]=(_Float16)p0.x;  r[1]=(_Float16)p0.y;  r[2]=(_Float16)p0.z;  r[3]=(_Float16)p0.w;
  r[4]=(_Float16)p1.x;  r[5]=(_Float16)p1.y;  r[6]=(_Float16)p1.z;  r[7]=(_Float16)p1.w;
  r[8]=(_Float16)p2.x;  r[9]=(_Float16)p2.y;  r[10]=(_Float16)p2.z; r[11]=(_Float16)p2.w;
  r[12]=(_Float16)p3.x; r[13]=(_Float16)p3.y; r[14]=(_Float16)p3.z; r[15]=(_Float16)p3.w;
  return r;
}

template<int EPI>
__global__ __launch_bounds__(WPB * 32)
void gemm_nt(const float* __restrict__ A, int lda,
             const _Float16* __restrict__ W16,
             const float* __restrict__ bias,
             float* __restrict__ C,
             int M, int K, float slope)
{
  const int lane = threadIdx.x & 31;
  const int wv   = threadIdx.x >> 5;
  const int m0   = (blockIdx.x * WPB + wv) * 16;
  if (m0 >= M) return;                     // wave-uniform: EXEC stays all-ones
  const int l16  = lane & 15;
  const int hi   = lane >> 4;              // 0: lanes 0-15, 1: lanes 16-31
  const int mrow = m0 + l16;
  const float* ap = A + (size_t)mrow * lda + hi * 16;         // A K-base 0 / 16
  const _Float16* wp = W16 + (size_t)l16 * K + hi * 16;       // B K-base 0 / 16

  v8f acc[8];
#pragma unroll
  for (int n = 0; n < 8; ++n) acc[n] = (v8f){0.f,0.f,0.f,0.f,0.f,0.f,0.f,0.f};

  for (int k0 = 0; k0 < K; k0 += 32) {
    // A 16x32 frag: lane<16 holds K {0..15}, lane>=16 {16..31}
    v16h a = pack16(*(const float4*)(ap + k0),
                    *(const float4*)(ap + k0 + 4),
                    *(const float4*)(ap + k0 + 8),
                    *(const float4*)(ap + k0 + 12));
    if (k0 + 32 < K) __builtin_prefetch(ap + k0 + 32, 0, 3);  // global_prefetch
#pragma unroll
    for (int n = 0; n < 8; ++n) {
      // B 32x16 frag for N-tile n: 16 contiguous halves of column n0+l16
      v16h b = *(const v16h*)(wp + (size_t)n * 16 * K + k0);
      acc[n] = __builtin_amdgcn_wmma_f32_16x16x32_f16(false, a, false, b,
                                                      (short)0, acc[n], false, false);
    }
  }
#pragma unroll
  for (int n = 0; n < 8; ++n) {
    const int ncol = n * 16 + l16;
    const float bb = bias ? bias[ncol] : 0.f;
#pragma unroll
    for (int r = 0; r < 8; ++r) {
      const int mr = m0 + (hi ? 8 + r : r);   // C layout: VGPR r -> M = r / 8+r
      float v = acc[n][r] + bb;
      if (EPI == 1) v = (v > 0.f) ? v : slope * v;
      if (EPI == 2) v = 1.f / (1.f + expf(-v));
      C[(size_t)mr * 128 + ncol] = v;
    }
  }
}

// fp32 [128 x K] (row stride ldw) -> compact f16 [128 x K]
__global__ void k_w2h(const float* __restrict__ W, int ldw,
                      _Float16* __restrict__ Wh, int K) {
  for (size_t i = (size_t)blockIdx.x*blockDim.x+threadIdx.x; i < (size_t)128*K;
       i += (size_t)gridDim.x*blockDim.x) {
    size_t r = i / (size_t)K; size_t c = i - r * (size_t)K;
    Wh[i] = (_Float16)W[r * (size_t)ldw + c];
  }
}

// ===========================================================================
// KNN: brute force per node (block/node), 24x argmin with index tie-break.
// ===========================================================================
__global__ __launch_bounds__(256)
void knn_kernel(const float* __restrict__ x, const int* __restrict__ batch,
                int* __restrict__ rowI, int n)
{
  __shared__ float d2s[2048];
  __shared__ float bd[256];
  __shared__ int   bi[256];
  const int i = blockIdx.x;
  const int t = threadIdx.x;
  const float xi0 = x[3*i], xi1 = x[3*i+1], xi2 = x[3*i+2];
  const int   bat = batch[i];
  for (int j = t; j < n; j += blockDim.x) {
    float dx = x[3*j]   - xi0;
    float dy = x[3*j+1] - xi1;
    float dz = x[3*j+2] - xi2;
    bool valid = (batch[j] == bat) && (j != i);
    d2s[j] = valid ? (dx*dx + dy*dy + dz*dz) : 3.0e38f;
  }
  __syncthreads();
  for (int kk = 0; kk < KNB; ++kk) {
    float mbest = 3.0e38f; int jbest = n;
    for (int j = t; j < n; j += blockDim.x) {
      float d = d2s[j];
      if (d < mbest || (d == mbest && j < jbest)) { mbest = d; jbest = j; }
    }
    bd[t] = mbest; bi[t] = jbest; __syncthreads();
    for (int s = 128; s > 0; s >>= 1) {
      if (t < s) {
        if (bd[t+s] < bd[t] || (bd[t+s] == bd[t] && bi[t+s] < bi[t])) {
          bd[t] = bd[t+s]; bi[t] = bi[t+s];
        }
      }
      __syncthreads();
    }
    if (t == 0) { rowI[i * KNB + kk] = bi[0]; d2s[bi[0]] = 3.0e38f; }
    __syncthreads();
  }
}

// ===========================================================================
// Per-edge static features: gaussian smearing + one-hot, ev_unit, u, ann.
// ===========================================================================
__global__ __launch_bounds__(128)
void edge_feat_kernel(const float* __restrict__ x, const int* __restrict__ rowI,
                      float* __restrict__ esf, float* __restrict__ evu,
                      float* __restrict__ u, float* __restrict__ ann)
{
  const int e = blockIdx.x, c = threadIdx.x;
  const int i = e / KNB, j = rowI[e];
  __shared__ float sh[4];
  if (c < 3) sh[c] = x[3*j + c] - x[3*i + c];
  __syncthreads();
  if (c == 0) sh[3] = sqrtf(sh[0]*sh[0] + sh[1]*sh[1] + sh[2]*sh[2]);
  __syncthreads();
  const float dist = sh[3];
  if (c < 3) evu[(size_t)e*3 + c] = sh[c] / (dist + 1e-7f);
  if (c == 0) {
    u[e] = dist / (dist + 1e-7f);
    float a = 0.5f * (cosf(dist * 3.14159265358979f / RMAXC) + 1.f);
    ann[e] = (dist <= RMAXC && dist >= 0.f) ? a : 0.f;
  }
  float val;
  if (c < NGC) {
    float off = RMAXC * (float)c / (float)(NGC - 1);
    float tdd = dist - off;
    const float coeff = -0.5f * ((float)(NGC-1)/RMAXC) * ((float)(NGC-1)/RMAXC);
    val = expf(coeff * tdd * tdd);
  } else {
    val = (c == NGC) ? 1.f : 0.f;     // one_hot(edge_type=0)
  }
  esf[(size_t)e * 128 + c] = val;
}

// ===========================================================================
// Elementwise kernels (grid-stride)
// ===========================================================================
#define GS_LOOP(i, n) for (size_t i = (size_t)blockIdx.x*blockDim.x+threadIdx.x; \
                           i < (n); i += (size_t)gridDim.x*blockDim.x)

__global__ void k_prep(const float* __restrict__ h, const float* __restrict__ vec,
                       float* hb, float* v0, float* v1, float* v2, size_t NF) {
  GS_LOOP(i, NF) {
    hb[i] = h[i];
    size_t b = i * 3;                 // (n*128+c)*3
    v0[i] = vec[b]; v1[i] = vec[b+1]; v2[i] = vec[b+2];
  }
}

__global__ void k_out(const float* __restrict__ hb, const float* __restrict__ v0,
                      const float* __restrict__ v1, const float* __restrict__ v2,
                      float* __restrict__ out, size_t NF) {
  GS_LOOP(i, NF) {
    out[i] = hb[i];
    size_t b = NF + i * 3;
    out[b] = v0[i]; out[b+1] = v1[i]; out[b+2] = v2[i];
  }
}

// wv1 = lv_w @ ve_w ; wv2 = lv2_w @ wv1 ; avec[s] = ls_w[s, 0:128] . |wv1|
__global__ __launch_bounds__(128)
void k_wprep(const float* __restrict__ lvw, const float* __restrict__ lv2w,
             const float* __restrict__ lsw, const float* __restrict__ vew,
             float* wv1, float* wv2, float* avec) {
  __shared__ float s1[128];
  const int t = threadIdx.x;
  float a = 0.f;
  for (int i = 0; i < 128; ++i) a += lvw[t*128 + i] * vew[i];
  s1[t] = a; wv1[t] = a;
  __syncthreads();
  float a2 = 0.f, a3 = 0.f;
  for (int hh = 0; hh < 128; ++hh) {
    a2 += lv2w[t*128 + hh] * s1[hh];
    a3 += lsw[t*256 + hh] * fabsf(s1[hh]);
  }
  wv2[t] = a2; avec[t] = a3;
}

// espre += u[e]*avec[s] (in place); es = leaky(espre, 0.01)
__global__ void k_espre_es(float* __restrict__ espre, float* __restrict__ es,
                           const float* __restrict__ u, const float* __restrict__ avec,
                           size_t EF) {
  GS_LOOP(i, EF) {
    size_t e = i >> 7; int s = (int)(i & 127);
    float v = espre[i] + u[e] * avec[s];
    espre[i] = v;
    es[i] = (v > 0.f) ? v : 0.01f * v;
  }
}

// gate -> gv = gate * wv2[c] (in place)
__global__ void k_gv(float* __restrict__ g, const float* __restrict__ wv2, size_t EF) {
  GS_LOOP(i, EF) g[i] *= wv2[i & 127];
}

// rank-1 vn_leaky_relu on edge coefficients
__global__ void k_f(const float* __restrict__ gv, const float* __restrict__ q,
                    const float* __restrict__ u, float* __restrict__ f, size_t EF) {
  GS_LOOP(i, EF) {
    size_t e = i >> 7;
    float ue = u[e]; float s = ue * ue;
    float x = gv[i], d = q[i];
    float dot = x * d * s;
    float dsq = d * d * s;
    float neg = x - (dot / (dsq + 1e-6f)) * d;
    f[i] = 0.2f * x + 0.8f * ((dot >= 0.f) ? x : neg);
  }
}

// y_sca and y_vec(3 planes) from edge/node coefficient arrays
__global__ void k_ybuild(const float* __restrict__ sct, const float* __restrict__ c1,
                         const float* __restrict__ g2,  const float* __restrict__ nsca,
                         const float* __restrict__ nv0, const float* __restrict__ nv1,
                         const float* __restrict__ nv2, const float* __restrict__ n2eo,
                         const float* __restrict__ evu, const float* __restrict__ evnb,
                         float* __restrict__ ysca, float* __restrict__ y0,
                         float* __restrict__ y1,  float* __restrict__ y2, size_t EF) {
  GS_LOOP(i, EF) {
    size_t e = i >> 7; int c = (int)(i & 127);
    size_t nidx = (e / KNB) * 128 + c;      // gather via col = e/K
    float ns  = nsca[nidx];
    float c2  = n2eo[nidx];
    float c1v = c1[i], gv2 = g2[i], bb = evnb[c];
    ysca[i] = ns * sct[i];
    float u0 = evu[e*3], u1 = evu[e*3+1], u2 = evu[e*3+2];
    y0[i] = c1v * nv0[nidx] + c2 * (gv2 * u0 + bb);
    y1[i] = c1v * nv1[nidx] + c2 * (gv2 * u1 + bb);
    y2[i] = c1v * nv2[nidx] + c2 * (gv2 * u2 + bb);
  }
}

// cat[r, 0:128] = ||vhid||_d ; cat[r, 128:256] = sca
__global__ void k_catnorm(const float* __restrict__ h0, const float* __restrict__ h1,
                          const float* __restrict__ h2, const float* __restrict__ sca,
                          float* __restrict__ cat, size_t MF) {
  GS_LOOP(i, MF) {
    size_t r = i >> 7; int c = (int)(i & 127);
    float a = h0[i], b = h1[i], d = h2[i];
    cat[r*256 + c]       = sqrtf(a*a + b*b + d*d);
    cat[r*256 + 128 + c] = sca[i];
  }
}

// out_d = gate * in_d (in place allowed)
__global__ void k_mulgate3(const float* __restrict__ g, float* v0, float* v1, float* v2,
                           size_t MF) {
  GS_LOOP(i, MF) { float gg = g[i]; v0[i] *= gg; v1[i] *= gg; v2[i] *= gg; }
}

// ms = msca*ann ; mv_d = gate*mvec_d*ann (in place)
__global__ void k_msg(const float* __restrict__ gate, const float* __restrict__ ann,
                      float* __restrict__ ms, float* v0, float* v1, float* v2, size_t EF) {
  GS_LOOP(i, EF) {
    float a = ann[i >> 7];
    ms[i] *= a;
    float ga = gate[i] * a;
    v0[i] *= ga; v1[i] *= ga; v2[i] *= ga;
  }
}

// segment_sum over row via atomics
__global__ void k_scatter(const int* __restrict__ rowI, const float* __restrict__ ms,
                          const float* __restrict__ v0, const float* __restrict__ v1,
                          const float* __restrict__ v2, float* __restrict__ as,
                          float* __restrict__ a0, float* __restrict__ a1,
                          float* __restrict__ a2, size_t EF) {
  GS_LOOP(i, EF) {
    size_t e = i >> 7; int c = (int)(i & 127);
    size_t d = (size_t)rowI[e] * 128 + c;
    atomicAdd(&as[d], ms[i]);
    atomicAdd(&a0[d], v0[i]);
    atomicAdd(&a1[d], v1[i]);
    atomicAdd(&a2[d], v2[i]);
  }
}

// scalar layernorm + leaky_relu (block per node)
__global__ __launch_bounds__(128)
void k_lnsca(const float* __restrict__ cs, const float* __restrict__ as,
             const float* __restrict__ g, const float* __restrict__ b,
             float* __restrict__ out) {
  __shared__ float red[128];
  const size_t n = blockIdx.x; const int c = threadIdx.x;
  float x = cs[n*128 + c] + as[n*128 + c];
  red[c] = x; __syncthreads();
  for (int s = 64; s > 0; s >>= 1) { if (c < s) red[c] += red[c+s]; __syncthreads(); }
  float mu = red[0] / 128.f; __syncthreads();
  float d = x - mu; red[c] = d*d; __syncthreads();
  for (int s = 64; s > 0; s >>= 1) { if (c < s) red[c] += red[c+s]; __syncthreads(); }
  float var = red[0] / 128.f;
  float y = d * rsqrtf(var + 1e-5f) * g[c] + b[c];
  out[n*128 + c] = (y > 0.f) ? y : 0.01f * y;
}

// vector layernorm over (NV,3) per node (block per node)
__global__ __launch_bounds__(128)
void k_lnvec(const float* __restrict__ c0, const float* __restrict__ c1,
             const float* __restrict__ c2, const float* __restrict__ a0,
             const float* __restrict__ a1, const float* __restrict__ a2,
             const float* __restrict__ g, const float* __restrict__ b,
             float* __restrict__ o0, float* __restrict__ o1, float* __restrict__ o2) {
  __shared__ float red[128];
  const size_t n = blockIdx.x; const int c = threadIdx.x;
  const size_t i = n*128 + c;
  float x0 = c0[i] + a0[i], x1 = c1[i] + a1[i], x2 = c2[i] + a2[i];
  red[c] = x0 + x1 + x2; __syncthreads();
  for (int s = 64; s > 0; s >>= 1) { if (c < s) red[c] += red[c+s]; __syncthreads(); }
  float mu = red[0] / 384.f; __syncthreads();
  float d0 = x0-mu, d1 = x1-mu, d2 = x2-mu;
  red[c] = d0*d0 + d1*d1 + d2*d2; __syncthreads();
  for (int s = 64; s > 0; s >>= 1) { if (c < s) red[c] += red[c+s]; __syncthreads(); }
  float inv = rsqrtf(red[0] / 384.f + 1e-5f);
  o0[i] = d0 * inv * g[c*3]   + b[c*3];
  o1[i] = d1 * inv * g[c*3+1] + b[c*3+1];
  o2[i] = d2 * inv * g[c*3+2] + b[c*3+2];
}

// vn_leaky_relu on full node vectors given dmat_d = ov_d @ W^T
__global__ void k_vnout(const float* __restrict__ x0, const float* __restrict__ x1,
                        const float* __restrict__ x2, const float* __restrict__ m0,
                        const float* __restrict__ m1, const float* __restrict__ m2,
                        float* o0, float* o1, float* o2, size_t NF) {
  GS_LOOP(i, NF) {
    float a0=x0[i],a1=x1[i],a2=x2[i], b0=m0[i],b1=m1[i],b2=m2[i];
    float dot = a0*b0 + a1*b1 + a2*b2;
    float dsq = b0*b0 + b1*b1 + b2*b2;
    float cf = dot / (dsq + 1e-6f);
    bool pos = dot >= 0.f;
    o0[i] = 0.2f*a0 + 0.8f*(pos ? a0 : a0 - cf*b0);
    o1[i] = 0.2f*a1 + 0.8f*(pos ? a1 : a1 - cf*b1);
    o2[i] = 0.2f*a2 + 0.8f*(pos ? a2 : a2 - cf*b2);
  }
}

// h += dh ; vec_d += gate*dvec_d
__global__ void k_resid(float* __restrict__ hb, const float* __restrict__ dh,
                        const float* __restrict__ gate,
                        float* v0, float* v1, float* v2,
                        const float* d0, const float* d1, const float* d2, size_t NF) {
  GS_LOOP(i, NF) {
    hb[i] += dh[i];
    float g = gate[i];
    v0[i] += g * d0[i]; v1[i] += g * d1[i]; v2[i] += g * d2[i];
  }
}

// ===========================================================================
// Host orchestration
// ===========================================================================
static void launch_gemm(hipStream_t st, int epi, const float* A, int lda,
                        const _Float16* W16, const float* bias,
                        float* C, int M, int K) {
  dim3 grid((M/16 + WPB - 1) / WPB), blk(WPB * 32);
  if (epi == 2)      gemm_nt<2><<<grid, blk, 0, st>>>(A, lda, W16, bias, C, M, K, 0.f);
  else if (epi == 1) gemm_nt<1><<<grid, blk, 0, st>>>(A, lda, W16, bias, C, M, K, 0.01f);
  else               gemm_nt<0><<<grid, blk, 0, st>>>(A, lda, W16, bias, C, M, K, 0.f);
}

static inline dim3 ewg(size_t n) { return dim3((unsigned)((n + 255) / 256)); }

extern "C" void kernel_launch(void* const* d_in, const int* in_sizes, int n_in,
                              void* d_out, int out_size, void* d_ws, size_t ws_size,
                              hipStream_t stream)
{
  const int N  = in_sizes[0] / 3;
  const int E  = N * KNB;
  const size_t NF = (size_t)N * 128;
  const size_t EF = (size_t)E * 128;

  const float* x     = (const float*)d_in[0];
  const float* vecin = (const float*)d_in[1];
  const float* hin   = (const float*)d_in[2];
  const int*   batch = (const int*)  d_in[3];
  const float* lvw   = (const float*)d_in[4];
  const float* lv2w  = (const float*)d_in[5];
  const float* lsw   = (const float*)d_in[6];
  const float* gww   = (const float*)d_in[7];
  const float* gbb   = (const float*)d_in[8];
  const float* vew   = (const float*)d_in[9];
  const float* vnre  = (const float*)d_in[10];
  const float* vnro  = (const float*)d_in[11];
  const float* scaw  = (const float*)d_in[12];
  const float* scab  = (const float*)d_in[13];
  const float* e2nw  = (const float*)d_in[14];
  const float* e2nb  = (const float*)d_in[15];
  const float* n2ew  = (const float*)d_in[16];
  const float* n2eb  = (const float*)d_in[17];
  const float* evnw  = (const float*)d_in[18];
  const float* evnb  = (const float*)d_in[19];
  const float* lnsg  = (const float*)d_in[20];
  const float* lnsb  = (const float*)d_in[21];
  const float* lnvg  = (const float*)d_in[22];
  const float* lnvb  = (const float*)d_in[23];

  auto LV  = [&](int l, int i){ return lvw  + (size_t)(l*5+i)*NVC*NVC; };
  auto LV2 = [&](int l, int i){ return lv2w + (size_t)(l*5+i)*NVC*NVC; };
  auto LS  = [&](int l, int i){ return lsw  + (size_t)(l*5+i)*NSC*(NSC+NVC); };
  auto GW  = [&](int l, int i){ return gww  + (size_t)(l*5+i)*NVC*NSC; };
  auto GB  = [&](int l, int i){ return gbb  + (size_t)(l*5+i)*NVC; };

  // ---- workspace carve-out (fp32 elements; every block multiple of 128) ----
  float* base = (float*)d_ws; size_t off = 0;
  auto alloc = [&](size_t n){ float* p = base + off; off += n; return p; };
  int*   rowI = (int*)alloc((size_t)E);
  float* evu  = alloc((size_t)E * 3);
  float* u    = alloc((size_t)E);
  float* ann  = alloc((size_t)E);
  float* esf  = alloc(EF);                 // edge scalar features (persistent)
  float* hb   = alloc(NF);
  float* vT0  = alloc(NF); float* vT1 = alloc(NF); float* vT2 = alloc(NF);
  float* EP[12]; for (int i = 0; i < 12; ++i) EP[i] = alloc(EF);   // edge pool
  // node scratch
  float* vh0 = alloc(NF), *vh1 = alloc(NF), *vh2 = alloc(NF);
  float* ncat = alloc(2*NF);
  float* nsca = alloc(NF), *ngate = alloc(NF);
  float* nov0 = alloc(NF), *nov1 = alloc(NF), *nov2 = alloc(NF);
  float* n2eo = alloc(NF);
  float* aggr = alloc(4*NF);               // [as, av0, av1, av2] contiguous
  float* as = aggr, *av0 = aggr+NF, *av1 = aggr+2*NF, *av2 = aggr+3*NF;
  float* cs  = alloc(NF), *cgate = alloc(NF);
  float* cv0 = alloc(NF), *cv1 = alloc(NF), *cv2 = alloc(NF);
  float* losa = alloc(NF);
  float* on0 = alloc(NF), *on1 = alloc(NF), *on2 = alloc(NF);
  float* dm0 = alloc(NF), *dm1 = alloc(NF), *dm2 = alloc(NF);
  float* vn0 = alloc(NF), *vn1 = alloc(NF), *vn2 = alloc(NF);
  float* dsca = alloc(NF), *dgate = alloc(NF);
  float* dv0 = alloc(NF), *dv1 = alloc(NF), *dv2 = alloc(NF);
  float* wv1 = alloc(128), *wv2 = alloc(128), *avec = alloc(128);
  // f16 weight pool (per-layer, re-converted each layer): 1 MB
  _Float16* WHbase = (_Float16*)alloc(524288 / 2);
  size_t ho = 0;
  auto halloc = [&](size_t n){ _Float16* p = WHbase + ho; ho += n; return p; };
  _Float16 *hLV[5], *hLV2[5], *hGW[5], *hLS[5];
  for (int i = 0; i < 5; ++i) { hLV[i] = halloc(16384); hLV2[i] = halloc(16384);
                                hGW[i] = halloc(16384); hLS[i] = halloc(32768); }
  _Float16* hLSr  = halloc(16384);
  _Float16* hvnre = halloc(16384);
  _Float16* hvnro = halloc(16384);
  _Float16* hscaw = halloc(16384);
  _Float16* he2nw = halloc(16384);
  _Float16* hn2ew = halloc(16384);
  _Float16* hevnw = halloc(16384);
  (void)ws_size; (void)n_in; (void)out_size;

  auto cvtW = [&](const float* W, int ldw, _Float16* dst, int K) {
    k_w2h<<<ewg((size_t)128*K), 256, 0, stream>>>(W, ldw, dst, K);
  };

  // gv_linear on (sca, vec[3]) -> (osca raw, gate, ovec raw); M rows.
  auto run_gvl = [&](int l, int i, int M, const float* sca,
                     const float* a0, const float* a1, const float* a2,
                     float* h0, float* h1, float* h2, float* cat,
                     float* osca, float* ogate, float* o0, float* o1, float* o2) {
    launch_gemm(stream, 0, a0, 128, hLV[i], nullptr, h0, M, 128);
    launch_gemm(stream, 0, a1, 128, hLV[i], nullptr, h1, M, 128);
    launch_gemm(stream, 0, a2, 128, hLV[i], nullptr, h2, M, 128);
    k_catnorm<<<ewg((size_t)M*128), 256, 0, stream>>>(h0, h1, h2, sca, cat, (size_t)M*128);
    launch_gemm(stream, 0, cat, 256, hLS[i], nullptr, osca, M, 256);
    launch_gemm(stream, 2, osca, 128, hGW[i], GB(l,i), ogate, M, 128);
    launch_gemm(stream, 0, h0, 128, hLV2[i], nullptr, o0, M, 128);
    launch_gemm(stream, 0, h1, 128, hLV2[i], nullptr, o1, M, 128);
    launch_gemm(stream, 0, h2, 128, hLV2[i], nullptr, o2, M, 128);
  };

  // ---- graph construction + static edge features ----
  k_prep<<<ewg(NF), 256, 0, stream>>>(hin, vecin, hb, vT0, vT1, vT2, NF);
  knn_kernel<<<N, 256, 0, stream>>>(x, batch, rowI, N);
  edge_feat_kernel<<<E, 128, 0, stream>>>(x, rowI, esf, evu, u, ann);

  for (int l = 0; l < LAY; ++l) {
    // --- convert this layer's GEMM weights to compact f16 ---
    for (int i = 0; i < 5; ++i) {
      if (i != 1) {                       // i==1 handled analytically (rank-1)
        cvtW(LV(l,i), 128, hLV[i], 128);
        cvtW(LV2(l,i), 128, hLV2[i], 128);
        cvtW(LS(l,i), 256, hLS[i], 256);
      }
      cvtW(GW(l,i), 128, hGW[i], 128);
    }
    cvtW(LS(l,1) + 128, 256, hLSr, 128);  // right half of edge ls_w
    cvtW(vnre + (size_t)l*ECC*ECC, 128, hvnre, 128);
    cvtW(vnro + (size_t)l*NVC*NVC, 128, hvnro, 128);
    cvtW(scaw + (size_t)l*NSC*ECC, 128, hscaw, 128);
    cvtW(e2nw + (size_t)l*NVC*ECC, 128, he2nw, 128);
    cvtW(n2ew + (size_t)l*NVC*NSC, 128, hn2ew, 128);
    cvtW(evnw + (size_t)l*NVC*ECC, 128, hevnw, 128);

    // --- node branch: gvl(l,0) -> ns, nv(gated) ; n2e projection ---
    run_gvl(l, 0, N, hb, vT0, vT1, vT2, vh0, vh1, vh2, ncat,
            nsca, ngate, nov0, nov1, nov2);
    k_mulgate3<<<ewg(NF), 256, 0, stream>>>(ngate, nov0, nov1, nov2, NF);
    launch_gemm(stream, 0, nsca, 128, hn2ew, n2eb + (size_t)l*NVC, n2eo, N, 128);

    // --- edge branch (rank-1 collapsed) ---
    k_wprep<<<1, 128, 0, stream>>>(LV(l,1), LV2(l,1), LS(l,1),
                                   vew + (size_t)l*ECC, wv1, wv2, avec);
    // es_pre = esf @ ls_w[:,128:]^T  (+ u*avec) ; es = leaky(es_pre)
    launch_gemm(stream, 0, esf, 128, hLSr, nullptr, EP[0], E, 128);
    k_espre_es<<<ewg(EF), 256, 0, stream>>>(EP[0], EP[1], u, avec, EF);
    // gate -> gv -> q -> f
    launch_gemm(stream, 2, EP[0], 128, hGW[1], GB(l,1), EP[2], E, 128);
    k_gv<<<ewg(EF), 256, 0, stream>>>(EP[2], wv2, EF);
    launch_gemm(stream, 0, EP[2], 128, hvnre, nullptr, EP[3], E, 128);
    k_f<<<ewg(EF), 256, 0, stream>>>(EP[2], EP[3], u, EP[4], EF);
    // sca_t, c1 (from es), g2 (from f)
    launch_gemm(stream, 0, EP[1], 128, hscaw, scab + (size_t)l*NSC, EP[5], E, 128);
    launch_gemm(stream, 0, EP[1], 128, he2nw, e2nb + (size_t)l*NVC, EP[6], E, 128);
    launch_gemm(stream, 0, EP[4], 128, hevnw, nullptr, EP[7], E, 128);
    // y_sca (EP8), y_vec (EP9..11)
    k_ybuild<<<ewg(EF), 256, 0, stream>>>(EP[5], EP[6], EP[7], nsca,
                                          nov0, nov1, nov2, n2eo, evu,
                                          evnb + (size_t)l*NVC,
                                          EP[8], EP[9], EP[10], EP[11], EF);
    // --- message gvl(l,2): mhid in EP1..3, cat in EP4..5, msca EP6, gate EP0,
    //     mvec overwrites EP9..11 ---
    run_gvl(l, 2, E, EP[8], EP[9], EP[10], EP[11], EP[1], EP[2], EP[3], EP[4],
            EP[6], EP[0], EP[9], EP[10], EP[11]);
    k_msg<<<ewg(EF), 256, 0, stream>>>(EP[0], ann, EP[6], EP[9], EP[10], EP[11], EF);
    // --- aggregate (segment_sum over row) ---
    hipMemsetAsync(aggr, 0, 4 * NF * sizeof(float), stream);
    k_scatter<<<ewg(EF), 256, 0, stream>>>(rowI, EP[6], EP[9], EP[10], EP[11],
                                           as, av0, av1, av2, EF);
    // --- update branch: gvl(l,3), layernorms, vn-relu, gvl(l,4), residual ---
    run_gvl(l, 3, N, hb, vT0, vT1, vT2, vh0, vh1, vh2, ncat,
            cs, cgate, cv0, cv1, cv2);
    k_mulgate3<<<ewg(NF), 256, 0, stream>>>(cgate, cv0, cv1, cv2, NF);
    k_lnsca<<<N, 128, 0, stream>>>(cs, as, lnsg + (size_t)l*NSC,
                                   lnsb + (size_t)l*NSC, losa);
    k_lnvec<<<N, 128, 0, stream>>>(cv0, cv1, cv2, av0, av1, av2,
                                   lnvg + (size_t)l*NVC*3, lnvb + (size_t)l*NVC*3,
                                   on0, on1, on2);
    launch_gemm(stream, 0, on0, 128, hvnro, nullptr, dm0, N, 128);
    launch_gemm(stream, 0, on1, 128, hvnro, nullptr, dm1, N, 128);
    launch_gemm(stream, 0, on2, 128, hvnro, nullptr, dm2, N, 128);
    k_vnout<<<ewg(NF), 256, 0, stream>>>(on0, on1, on2, dm0, dm1, dm2,
                                         vn0, vn1, vn2, NF);
    run_gvl(l, 4, N, losa, vn0, vn1, vn2, vh0, vh1, vh2, ncat,
            dsca, dgate, dv0, dv1, dv2);
    k_resid<<<ewg(NF), 256, 0, stream>>>(hb, dsca, dgate, vT0, vT1, vT2,
                                         dv0, dv1, dv2, NF);
  }

  k_out<<<ewg(NF), 256, 0, stream>>>(hb, vT0, vT1, vT2, (float*)d_out, NF);
}
